// GATNet_2_44281112822527
// MI455X (gfx1250) — matrix operand order repaired and checked by reading, not compiled, not run on hardware
//
#include <hip/hip_runtime.h>
#include <hip/hip_bf16.h>

typedef _Float16 h8   __attribute__((ext_vector_type(8)));
typedef _Float16 v16h __attribute__((ext_vector_type(16)));
typedef float    v8f  __attribute__((ext_vector_type(8)));

#define NEG_SLOPE 0.2f

// ---------------------------------------------------------------------------
// Prep: x [N,16] f32 -> xh [N,32] f16 (zero-padded K)
// ---------------------------------------------------------------------------
__global__ void prep_x_kernel(const float* __restrict__ x, _Float16* __restrict__ xh, int N) {
    int n = blockIdx.x;
    int t = threadIdx.x;                       // blockDim = 32
    if (n >= N) return;
    xh[(size_t)n * 32 + t] = (t < 16) ? (_Float16)x[(size_t)n * 16 + t] : (_Float16)0.f;
}

// Prep: W [K,Nout] f32 row-major -> Wt [Ntp,Kp] f16 (transposed, zero-padded)
__global__ void prep_wt_kernel(const float* __restrict__ W, _Float16* __restrict__ Wt,
                               int K, int Nout, int Kp, int Ntp) {
    int idx = blockIdx.x * blockDim.x + threadIdx.x;
    if (idx >= Ntp * Kp) return;
    int nn = idx / Kp, kk = idx % Kp;
    float v = (nn < Nout && kk < K) ? W[(size_t)kk * Nout + nn] : 0.f;
    Wt[idx] = (_Float16)v;
}

// ---------------------------------------------------------------------------
// WMMA GEMM: out[M,Nout](f16, ldo, +colOff) = act( A[M,Kp](f16) @ Wt[Ntp,Kp]^T + bias )
// One wave per 16x16 C tile, K-loop step 32, v_wmma_f32_16x16x32_f16.
// Lane layouts per CDNA5 ISA 7.12.2 (wave32).
// ---------------------------------------------------------------------------
__global__ __launch_bounds__(256)
void wmma_gemm_kernel(const _Float16* __restrict__ A, const _Float16* __restrict__ Wt,
                      const float* __restrict__ bias, _Float16* __restrict__ out,
                      int Kp, int ldo, int colOff, int M, int Nout, int Ntiles, int doBR)
{
    int wave = threadIdx.x >> 5;
    int lane = threadIdx.x & 31;
    int nt   = blockIdx.x * 8 + wave;          // N tile handled by this wave
    if (nt >= Ntiles) return;                  // uniform per wave -> EXEC stays all-1s
    int mt = blockIdx.y;
    int r  = lane & 15;                        // row (A) / col (B,C) within tile
    int hf = lane >> 4;                        // lane half selector

    const _Float16* Arow = A  + (size_t)(mt * 16 + r) * Kp;
    int ncol             = nt * 16 + r;
    const _Float16* Wrow = Wt + (size_t)ncol * Kp;

    v8f acc = {};
    for (int kb = 0; kb < Kp; kb += 32) {
        // A 16x32 f16: lane<16 -> K 0..7 & 16..23 ; lane>=16 -> K 8..15 & 24..31
        h8 a0 = *(const h8*)(Arow + kb + hf * 8);
        h8 a1 = *(const h8*)(Arow + kb + 16 + hf * 8);
        // B 32x16 f16 (from transposed W): lane<16 -> K 0..7 & 16..23 ; lane>=16 -> K 8..15 & 24..31
        h8 b0 = *(const h8*)(Wrow + kb + hf * 8);
        h8 b1 = *(const h8*)(Wrow + kb + 16 + hf * 8);
        v16h a, b;
#pragma unroll
        for (int i = 0; i < 8; ++i) {
            a[i] = a0[i]; a[8 + i] = a1[i];
            b[i] = b0[i]; b[8 + i] = b1[i];
        }
        acc = __builtin_amdgcn_wmma_f32_16x16x32_f16(false, a, false, b,
                                                     (short)0, acc, false, false);
    }

    if (ncol < Nout) {
        float bv = doBR ? bias[ncol] : 0.f;
#pragma unroll
        for (int v = 0; v < 8; ++v) {
            int row = mt * 16 + hf * 8 + v;    // C layout: VGPR v, lane-half hf
            float val = acc[v];
            if (doBR) { val += bv; val = fmaxf(val, 0.f); }
            if (row < M)
                out[(size_t)row * ldo + colOff + ncol] = (_Float16)val;
        }
    }
}

// ---------------------------------------------------------------------------
// Attention logits: al_src/al_dst [N,H] from xl (f16, ld=H*C)
// ---------------------------------------------------------------------------
__global__ void compute_al_kernel(const _Float16* __restrict__ xl, int ld,
                                  const float* __restrict__ a_src, const float* __restrict__ a_dst,
                                  float* __restrict__ alsrc, float* __restrict__ aldst,
                                  int N, int H, int C) {
    int i = blockIdx.x * blockDim.x + threadIdx.x;
    if (i >= N * H) return;
    int n = i / H, h = i % H;
    const _Float16* p = xl + (size_t)n * ld + h * C;
    float s = 0.f, d = 0.f;
    for (int c = 0; c < C; ++c) {
        float v = (float)p[c];
        s += v * a_src[h * C + c];
        d += v * a_dst[h * C + c];
    }
    alsrc[i] = s;
    aldst[i] = d;
}

// ---------------------------------------------------------------------------
// Softmax denominator per (dst, head). den must be zeroed first.
// Edges e<E come from edge_index; e>=E are self-loops (node e-E).
// ---------------------------------------------------------------------------
__global__ void edge_den_kernel(const long long* __restrict__ ei, int E, int N,
                                const float* __restrict__ alsrc, const float* __restrict__ aldst,
                                float* __restrict__ den, int H) {
    int e = blockIdx.x * blockDim.x + threadIdx.x;
    int Etot = E + N;
    if (e >= Etot) return;
    int s, d;
    if (e < E) { s = (int)ei[e]; d = (int)ei[(size_t)E + e]; }
    else       { s = e - E; d = e - E; }
    for (int h = 0; h < H; ++h) {
        float sc = alsrc[s * H + h] + aldst[d * H + h];
        sc = (sc > 0.f) ? sc : NEG_SLOPE * sc;
        atomicAdd(&den[d * H + h], __expf(sc));
    }
}

// ---------------------------------------------------------------------------
// Weighted scatter: acc[dst, :] += coef[h] * xl[src, :]. acc zeroed first.
// One block per edge; coef per head staged in LDS.
// ---------------------------------------------------------------------------
__global__ __launch_bounds__(256)
void edge_agg_kernel(const long long* __restrict__ ei, int E, int N,
                     const float* __restrict__ alsrc, const float* __restrict__ aldst,
                     const float* __restrict__ den,
                     const _Float16* __restrict__ xl, float* __restrict__ acc,
                     int H, int C) {
    __shared__ float coef[16];
    int e = blockIdx.x;
    int s, d;
    if (e < E) { s = (int)ei[e]; d = (int)ei[(size_t)E + e]; }
    else       { s = e - E; d = e - E; }
    int t = threadIdx.x;
    if (t < H) {
        float sc = alsrc[s * H + t] + aldst[d * H + t];
        sc = (sc > 0.f) ? sc : NEG_SLOPE * sc;
        coef[t] = __expf(sc) / den[d * H + t];
    }
    __syncthreads();
    int HC = H * C;
    const _Float16* xs = xl + (size_t)s * HC;
    float* ad = acc + (size_t)d * HC;
    for (int i = t; i < HC; i += blockDim.x)
        atomicAdd(&ad[i], coef[i / C] * (float)xs[i]);
}

// ---------------------------------------------------------------------------
// out[n, colOff + c] (f16, ldo) = relu(acc[n,c] + bias[c])
// ---------------------------------------------------------------------------
__global__ void bias_relu_h_kernel(const float* __restrict__ acc, const float* __restrict__ bias,
                                   _Float16* __restrict__ out, int ldo, int colOff,
                                   int HC, int N) {
    int i = blockIdx.x * blockDim.x + threadIdx.x;
    if (i >= N * HC) return;
    int n = i / HC, c = i % HC;
    float v = acc[(size_t)i] + bias[c];
    out[(size_t)n * ldo + colOff + c] = (_Float16)fmaxf(v, 0.f);
}

// ---------------------------------------------------------------------------
// Final GEMV [N,64] @ [64,1] + bias, sigmoid
// ---------------------------------------------------------------------------
__global__ void final_out_kernel(const _Float16* __restrict__ t2, const float* __restrict__ w,
                                 const float* __restrict__ b, float* __restrict__ out, int N) {
    int n = blockIdx.x * blockDim.x + threadIdx.x;
    if (n >= N) return;
    float s = b[0];
#pragma unroll
    for (int i = 0; i < 64; ++i) s += (float)t2[(size_t)n * 64 + i] * w[i];
    out[n] = 1.f / (1.f + __expf(-s));
}

// ---------------------------------------------------------------------------
// Host side
// ---------------------------------------------------------------------------
static inline size_t alignup(size_t x) { return (x + 255) & ~(size_t)255; }

static void run_gemm(hipStream_t st, const _Float16* A, int Kp, const _Float16* Wt,
                     const float* bias, _Float16* out, int ldo, int colOff,
                     int M, int Nout, int doBR) {
    int ntiles = (Nout + 15) / 16;
    dim3 grid((ntiles + 7) / 8, M / 16);
    wmma_gemm_kernel<<<grid, 256, 0, st>>>(A, Wt, bias, out, Kp, ldo, colOff,
                                           M, Nout, ntiles, doBR);
}

extern "C" void kernel_launch(void* const* d_in, const int* in_sizes, int n_in,
                              void* d_out, int out_size, void* d_ws, size_t ws_size,
                              hipStream_t stream) {
    const int N = 50000, E = 400000, Etot = E + N;

    const float*     x    = (const float*)d_in[0];
    const long long* ei   = (const long long*)d_in[1];
    const float *W1 = (const float*)d_in[2],  *as1 = (const float*)d_in[3],
                *ad1 = (const float*)d_in[4], *b1 = (const float*)d_in[5];
    const float *W2 = (const float*)d_in[6],  *as2 = (const float*)d_in[7],
                *ad2 = (const float*)d_in[8], *b2 = (const float*)d_in[9];
    const float *W3 = (const float*)d_in[10], *as3 = (const float*)d_in[11],
                *ad3 = (const float*)d_in[12], *b3 = (const float*)d_in[13];
    const float *ln1_w = (const float*)d_in[14], *ln1_b = (const float*)d_in[15];
    const float *ln2_w = (const float*)d_in[16], *ln2_b = (const float*)d_in[17];
    const float *la1_w = (const float*)d_in[18], *la1_b = (const float*)d_in[19];
    const float *la3_w = (const float*)d_in[20], *la3_b = (const float*)d_in[21];
    const float *lf1_w = (const float*)d_in[22], *lf1_b = (const float*)d_in[23];
    const float *lf2_w = (const float*)d_in[24], *lf2_b = (const float*)d_in[25];
    const float *lf3_w = (const float*)d_in[26], *lf3_b = (const float*)d_in[27];

    // ---- workspace layout (all offsets 256B aligned) ----
    char* ws = (char*)d_ws;
    size_t off = 0;
    auto alloc = [&](size_t bytes) -> char* {
        char* p = ws + off;
        off = alignup(off + bytes);
        return p;
    };
    _Float16* xh   = (_Float16*)alloc((size_t)N * 32 * 2);
    _Float16* w1t  = (_Float16*)alloc((size_t)256 * 32 * 2);
    _Float16* w2t  = (_Float16*)alloc((size_t)256 * 256 * 2);
    _Float16* w3t  = (_Float16*)alloc((size_t)768 * 256 * 2);
    _Float16* la1t = (_Float16*)alloc((size_t)80 * 256 * 2);
    _Float16* la3t = (_Float16*)alloc((size_t)208 * 256 * 2);
    _Float16* ln1t = (_Float16*)alloc((size_t)32 * 32 * 2);
    _Float16* ln2t = (_Float16*)alloc((size_t)32 * 32 * 2);
    _Float16* lf1t = (_Float16*)alloc((size_t)208 * 1088 * 2);
    _Float16* lf2t = (_Float16*)alloc((size_t)64 * 224 * 2);
    _Float16* xl   = (_Float16*)alloc((size_t)N * 768 * 2);   // per-layer x@W (reused)
    _Float16* x1h  = (_Float16*)alloc((size_t)N * 256 * 2);
    _Float16* x2h  = (_Float16*)alloc((size_t)N * 256 * 2);
    _Float16* g1h  = (_Float16*)alloc((size_t)N * 32 * 2);
    _Float16* xf   = (_Float16*)alloc((size_t)N * 1088 * 2);  // concat [gg|x3|xa1|xa2] + pad
    _Float16* t1   = (_Float16*)alloc((size_t)N * 224 * 2);
    _Float16* t2   = (_Float16*)alloc((size_t)N * 64 * 2);
    float*    acc   = (float*)alloc((size_t)N * 768 * 4);     // scatter accumulator (reused)
    float*    alsrc = (float*)alloc((size_t)N * 12 * 4);
    float*    aldst = (float*)alloc((size_t)N * 12 * 4);
    float*    den   = (float*)alloc((size_t)N * 12 * 4);

    // ---- prep (f16 conversions / transposes) ----
    prep_x_kernel<<<N, 32, 0, stream>>>(x, xh, N);
    auto prep = [&](const float* W, _Float16* Wt, int K, int Nout, int Kp, int Ntp) {
        int tot = Ntp * Kp;
        prep_wt_kernel<<<(tot + 255) / 256, 256, 0, stream>>>(W, Wt, K, Nout, Kp, Ntp);
    };
    prep(W1, w1t, 16, 256, 32, 256);
    prep(W2, w2t, 256, 256, 256, 256);
    prep(W3, w3t, 256, 768, 256, 768);
    prep(la1_w, la1t, 256, 80, 256, 80);
    prep(la3_w, la3t, 256, 200, 256, 208);
    prep(ln1_w, ln1t, 16, 32, 32, 32);
    prep(ln2_w, ln2t, 32, 32, 32, 32);
    prep(lf1_w, lf1t, 1080, 200, 1088, 208);
    prep(lf2_w, lf2t, 200, 64, 224, 64);

    hipMemsetAsync(xf, 0, (size_t)N * 1088 * 2, stream);   // zero concat pad cols
    hipMemsetAsync(t1, 0, (size_t)N * 224 * 2, stream);    // zero t1 pad cols

    // ---- dense gg path: gg = relu(relu(x@ln1+b)@ln2+b) -> xf cols [0,32) ----
    run_gemm(stream, xh, 32, ln1t, ln1_b, g1h, 32, 0, N, 32, 1);
    run_gemm(stream, g1h, 32, ln2t, ln2_b, xf, 1088, 0, N, 32, 1);

    // ---- GAT layer helper ----
    auto gat = [&](const _Float16* Ain, int Kp, const _Float16* Wt,
                   const float* a_s, const float* a_d, const float* bias,
                   int H, int C, _Float16* xout, int ldo, int colOff) {
        int HC = H * C;
        run_gemm(stream, Ain, Kp, Wt, nullptr, xl, HC, 0, N, HC, 0);
        int nh = N * H;
        compute_al_kernel<<<(nh + 255) / 256, 256, 0, stream>>>(xl, HC, a_s, a_d,
                                                                alsrc, aldst, N, H, C);
        hipMemsetAsync(den, 0, (size_t)N * H * 4, stream);
        edge_den_kernel<<<(Etot + 255) / 256, 256, 0, stream>>>(ei, E, N, alsrc, aldst, den, H);
        hipMemsetAsync(acc, 0, (size_t)N * HC * 4, stream);
        edge_agg_kernel<<<Etot, 256, 0, stream>>>(ei, E, N, alsrc, aldst, den, xl, acc, H, C);
        int tot = N * HC;
        bias_relu_h_kernel<<<(tot + 255) / 256, 256, 0, stream>>>(acc, bias, xout, ldo,
                                                                  colOff, HC, N);
    };

    // conv1: 16(32p) -> 8x32 ; conv2: 256 -> 8x32 ; conv3: 256 -> 12x64 (into xf cols 32..800)
    gat(xh, 32, w1t, as1, ad1, b1, 8, 32, x1h, 256, 0);
    gat(x1h, 256, w2t, as2, ad2, b2, 8, 32, x2h, 256, 0);
    gat(x2h, 256, w3t, as3, ad3, b3, 12, 64, xf, 1088, 32);

    // ---- side branches into concat buffer ----
    run_gemm(stream, x1h, 256, la1t, la1_b, xf, 1088, 800, N, 80, 1);   // xa1
    run_gemm(stream, x2h, 256, la3t, la3_b, xf, 1088, 880, N, 200, 1);  // xa2

    // ---- head ----
    run_gemm(stream, xf, 1088, lf1t, lf1_b, t1, 224, 0, N, 200, 1);
    run_gemm(stream, t1, 224, lf2t, lf2_b, t2, 64, 0, N, 64, 1);
    final_out_kernel<<<(N + 255) / 256, 256, 0, stream>>>(t2, lf3_w, lf3_b, (float*)d_out, N);

    (void)in_sizes; (void)n_in; (void)out_size; (void)ws_size;
}
